// GPSWithEdgeLayer_73632919322929
// MI455X (gfx1250) — matrix operand order repaired and checked by reading, not compiled, not run on hardware
//
#include <hip/hip_runtime.h>

#define DEV __device__ __forceinline__

typedef __attribute__((ext_vector_type(16))) __bf16 v16bf;
typedef __attribute__((ext_vector_type(8)))  __bf16 v8bf;
typedef __attribute__((ext_vector_type(8)))  float  v8f;
typedef __attribute__((ext_vector_type(4)))  float  v4f;

constexpr int Nn  = 32768;
constexpr int Ee  = 524288;
constexpr int Cc  = 128;
constexpr int Ll  = 1024;   // nodes per graph; G=32, H=4, DH=32

// ---------------- WMMA fragment helpers (CDNA5 16x16x32 bf16 layouts) -----------
// A (16x32 bf16): lane holds row=lane&15; kg=lane>>4. Lane's 16 elements are two
// contiguous K-runs of 8: [kg*8 .. kg*8+7] and [kg*8+16 .. kg*8+23].
// B (32x16 bf16): lane holds col=lane&15; kg=lane>>4, same K order as A:
// [kg*8 .. kg*8+7] and [kg*8+16 .. kg*8+23] of a K-major (transposed) matrix.
// C/D (16x16 f32): lane holds col=lane&15, rows (lane>>4)*8 + i.

DEV v16bf cat8(v8bf lo, v8bf hi) {
  return __builtin_shufflevector(lo, hi, 0, 1, 2, 3, 4, 5, 6, 7,
                                 8, 9, 10, 11, 12, 13, 14, 15);
}

// f32x8 -> bf16x8 (staging convert)
DEV v8bf pack_bf8(v4f a, v4f b) {
  v8bf o;
#pragma unroll
  for (int q = 0; q < 4; ++q) {
    o[q]     = (__bf16)a[q];
    o[4 + q] = (__bf16)b[q];
  }
  return o;
}

// A fragment from bf16 row-major tile (ld multiple of 8) -> 2x ds_load_b128
DEV v16bf load_a_bf16v(const __bf16* base, int ld, int kbase, int lane) {
  int row = lane & 15, kg = lane >> 4;
  const __bf16* p = base + row * ld + kbase + kg * 8;
  return cat8(*(const v8bf*)p, *(const v8bf*)(p + 16));
}

// A fragment from f32 row-major tile (ld multiple of 4) -> 4x ds_load_b128 + cvt
// (used only for the attention probability tile, which must stay f32 for softmax)
DEV v16bf load_a_f32v(const float* base, int ld, int kbase, int lane) {
  int row = lane & 15, kg = lane >> 4;
  const float* p = base + row * ld + kbase + kg * 8;
  v4f a0 = *(const v4f*)p,        a1 = *(const v4f*)(p + 4);
  v4f a2 = *(const v4f*)(p + 16), a3 = *(const v4f*)(p + 20);
  v16bf a;
#pragma unroll
  for (int i = 0; i < 4; ++i) {
    a[i]      = (__bf16)a0[i];
    a[4 + i]  = (__bf16)a1[i];
    a[8 + i]  = (__bf16)a2[i];
    a[12 + i] = (__bf16)a3[i];
  }
  return a;
}

// B fragment from a K-major (transposed) bf16 matrix Bt[n][k], ldk = K stride.
// Per-lane two 8-element K-runs (same order as A) -> 2x b128 loads (global or ds).
DEV v16bf load_b_t(const __bf16* Bt, int ldk, int kbase, int n0, int lane) {
  int col = lane & 15, kg = lane >> 4;
  const __bf16* p = Bt + (size_t)(n0 + col) * ldk + kbase + kg * 8;
  return cat8(*(const v8bf*)p, *(const v8bf*)(p + 16));
}

DEV v8f wmma_bf16(v16bf a, v16bf b, v8f c) {
  return __builtin_amdgcn_wmma_f32_16x16x32_bf16(false, a, false, b, (short)0, c,
                                                 false, false);
}

// ---------------- Kernel 0: fp32 -> bf16 transpose-convert (weights K-major) ----
__global__ void k_cvt_t(const float* __restrict__ s, __bf16* __restrict__ d,
                        int K, int Ncol) {
  int i = blockIdx.x * blockDim.x + threadIdx.x;
  if (i >= K * Ncol) return;
  int n = i / K, k = i - n * K;
  d[(size_t)n * K + k] = (__bf16)s[(size_t)k * Ncol + n];
}

// ---------------- Kernel 1: edge MLP + scatter-sum -------------------------------
__global__ __launch_bounds__(256) void k_edge_mlp(
    const float* __restrict__ x, const int* __restrict__ eidx,
    const float* __restrict__ eattr, const float* __restrict__ b_edge,
    const __bf16* __restrict__ WbT, float* __restrict__ e_new,
    float* __restrict__ h_conv) {
  __shared__ __bf16 As[16 * 392];         // 16 edges x 384 bf16 (padded ld)
  __shared__ int sidx[16], didx[16];
  const int tid = threadIdx.x;
  const int e0 = blockIdx.x * 16;
  if (tid < 16) {
    sidx[tid] = eidx[e0 + tid];
    didx[tid] = eidx[Ee + e0 + tid];
  }
  __syncthreads();
  for (int idx = tid; idx < 16 * 48; idx += 256) {   // 8 elems per step
    int j = idx / 48, s = idx - j * 48, k = s * 8;
    const float* src;
    if (k < 128)      src = x + (size_t)sidx[j] * Cc + k;
    else if (k < 256) src = x + (size_t)didx[j] * Cc + (k - 128);
    else              src = eattr + (size_t)(e0 + j) * Cc + (k - 256);
    *(v8bf*)(As + j * 392 + k) =
        pack_bf8(*(const v4f*)src, *(const v4f*)(src + 4));
  }
  __syncthreads();
  const int wave = tid >> 5, lane = tid & 31;
  const int n0 = wave * 16, col = lane & 15, rbase = (lane >> 4) * 8;
  v8f c;
  float bias = b_edge[n0 + col];
#pragma unroll
  for (int i = 0; i < 8; ++i) c[i] = bias;
#pragma unroll
  for (int kk = 0; kk < 12; ++kk) {
    v16bf a = load_a_bf16v(As, 392, kk * 32, lane);
    v16bf b = load_b_t(WbT, 384, kk * 32, n0, lane);
    c = wmma_bf16(a, b, c);
  }
#pragma unroll
  for (int i = 0; i < 8; ++i) {
    int m = rbase + i;
    float v = c[i] > 0.f ? c[i] : 0.f;
    e_new[(size_t)(e0 + m) * Cc + n0 + col] = v;
    atomicAdd(&h_conv[(size_t)didx[m] * Cc + n0 + col], v);
  }
}

// ---------------- Kernel 2: QKV projection (output bf16) -------------------------
__global__ __launch_bounds__(256) void k_qkv(
    const float* __restrict__ x, const float* __restrict__ bqkv,
    const __bf16* __restrict__ WbT, __bf16* __restrict__ qkv) {
  __shared__ __bf16 As[16 * 136];
  const int tid = threadIdx.x;
  const int m0 = blockIdx.x * 16;
  for (int idx = tid; idx < 16 * 16; idx += 256) {
    int j = idx >> 4, k = (idx & 15) * 8;
    const float* src = x + (size_t)(m0 + j) * Cc + k;
    *(v8bf*)(As + j * 136 + k) =
        pack_bf8(*(const v4f*)src, *(const v4f*)(src + 4));
  }
  __syncthreads();
  const int wave = tid >> 5, lane = tid & 31;
  const int n0 = blockIdx.y * 128 + wave * 16;
  const int col = lane & 15, rbase = (lane >> 4) * 8;
  v8f c;
  float bias = bqkv[n0 + col];
#pragma unroll
  for (int i = 0; i < 8; ++i) c[i] = bias;
#pragma unroll
  for (int kk = 0; kk < 4; ++kk) {
    v16bf a = load_a_bf16v(As, 136, kk * 32, lane);
    v16bf b = load_b_t(WbT, 128, kk * 32, n0, lane);
    c = wmma_bf16(a, b, c);
  }
#pragma unroll
  for (int i = 0; i < 8; ++i)
    qkv[(size_t)(m0 + rbase + i) * 384 + n0 + col] = (__bf16)c[i];
}

// ---------------- Kernel 3: per-(graph, head) attention ---------------------------
constexpr int SLD = 1028;   // f32 score tile leading dim (padded)
constexpr int VLD = 1032;   // bf16 V^T leading dim (16B-aligned rows, low conflict)
constexpr size_t A_QS  = 0;                            // 16x32 bf16    1024
constexpr size_t A_KS  = 1024;                         // 1024x32 bf16  65536
constexpr size_t A_VT  = A_KS + 65536;                 // 32x1032 bf16  66048
constexpr size_t A_S   = A_VT + 66048;                 // 16x1028 f32   65792
constexpr size_t A_RED = A_S + (size_t)16 * SLD * 4;   // 16x16 f32     1024
constexpr size_t A_RMX = A_RED + 1024;                 // 16 f32
constexpr size_t A_RSM = A_RMX + 64;                   // 16 f32
constexpr size_t A_OP  = A_RSM + 64;                   // 8x16x32 f32   16384
constexpr size_t A_TOT = A_OP + 16384;                 // = 215936 bytes

__global__ __launch_bounds__(256) void k_attn(const __bf16* __restrict__ qkv,
                                              float* __restrict__ attn_o) {
  extern __shared__ char smem[];
  __bf16* Qs = (__bf16*)(smem + A_QS);
  __bf16* Ks = (__bf16*)(smem + A_KS);
  __bf16* Vt = (__bf16*)(smem + A_VT);
  float*  S  = (float*)(smem + A_S);
  float*  red = (float*)(smem + A_RED);
  float*  rowmax = (float*)(smem + A_RMX);
  float*  rowsum = (float*)(smem + A_RSM);
  float*  opart  = (float*)(smem + A_OP);

  const int tid = threadIdx.x;
  const int blk = blockIdx.x;
  const int qt = blk & 63;
  const int h  = (blk >> 6) & 3;
  const int g  = blk >> 8;
  const size_t rowbase = (size_t)g * Ll;

  // stage K row-major, V transposed (d-major), Q row-major
  for (int r = tid; r < Ll; r += 256) {
    const __bf16* src = qkv + (rowbase + r) * 384 + h * 32;
    *(v8bf*)(Ks + r * 32)      = *(const v8bf*)(src + 128);
    *(v8bf*)(Ks + r * 32 + 8)  = *(const v8bf*)(src + 136);
    *(v8bf*)(Ks + r * 32 + 16) = *(const v8bf*)(src + 144);
    *(v8bf*)(Ks + r * 32 + 24) = *(const v8bf*)(src + 152);
    v8bf v0 = *(const v8bf*)(src + 256), v1 = *(const v8bf*)(src + 264);
    v8bf v2 = *(const v8bf*)(src + 272), v3 = *(const v8bf*)(src + 280);
#pragma unroll
    for (int d = 0; d < 8; ++d) {
      Vt[(d)      * VLD + r] = v0[d];
      Vt[(d + 8)  * VLD + r] = v1[d];
      Vt[(d + 16) * VLD + r] = v2[d];
      Vt[(d + 24) * VLD + r] = v3[d];
    }
  }
  for (int idx = tid; idx < 16 * 4; idx += 256) {
    int r = idx >> 2, sg = (idx & 3) * 8;
    *(v8bf*)(Qs + r * 32 + sg) =
        *(const v8bf*)(qkv + (rowbase + qt * 16 + r) * 384 + h * 32 + sg);
  }
  __syncthreads();

  const int wave = tid >> 5, lane = tid & 31;
  const int col = lane & 15, rbase = (lane >> 4) * 8;
  const float scale = 0.17677669529663687f;   // 1/sqrt(DH=32)

  // scores: q [16x32] @ K^T — each wave covers 128 key cols (8 tiles of 16)
  v16bf aq = load_a_bf16v(Qs, 32, 0, lane);
#pragma unroll
  for (int t = 0; t < 8; ++t) {
    int ct = wave * 8 + t;
    v16bf bk = load_b_t(Ks, 32, 0, ct * 16, lane);   // B(k,n) = K[ct*16+n][k]
    v8f c;
#pragma unroll
    for (int i = 0; i < 8; ++i) c[i] = 0.f;
    c = wmma_bf16(aq, bk, c);
#pragma unroll
    for (int i = 0; i < 8; ++i)
      S[(rbase + i) * SLD + ct * 16 + col] = c[i] * scale;
  }
  __syncthreads();

  // block-wide softmax: 16 threads per row, vectorized sweeps
  {
    int row = tid >> 4, t = tid & 15;
    float* Srow = S + row * SLD;
    float m = -3.4e38f;
    for (int jj = t; jj < 256; jj += 16) {
      v4f v = *(const v4f*)(Srow + jj * 4);
      m = fmaxf(fmaxf(fmaxf(m, v[0]), v[1]), fmaxf(v[2], v[3]));
    }
    red[row * 16 + t] = m;
    __syncthreads();
    if (t == 0) {
      float mm = red[row * 16];
#pragma unroll
      for (int j = 1; j < 16; ++j) mm = fmaxf(mm, red[row * 16 + j]);
      rowmax[row] = mm;
    }
    __syncthreads();
    float mm = rowmax[row];
    float s = 0.f;
    for (int jj = t; jj < 256; jj += 16) {
      v4f v = *(const v4f*)(Srow + jj * 4);
#pragma unroll
      for (int q = 0; q < 4; ++q) {
        v[q] = __expf(v[q] - mm);
        s += v[q];
      }
      *(v4f*)(Srow + jj * 4) = v;
    }
    red[row * 16 + t] = s;
    __syncthreads();
    if (t == 0) {
      float ss = 0.f;
#pragma unroll
      for (int j = 0; j < 16; ++j) ss += red[row * 16 + j];
      rowsum[row] = ss;
    }
    __syncthreads();
  }

  // o_partial = P[:, wave*128 .. +127] @ V  -> 16x32 per wave
  v8f o0, o1;
#pragma unroll
  for (int i = 0; i < 8; ++i) { o0[i] = 0.f; o1[i] = 0.f; }
#pragma unroll
  for (int t = 0; t < 4; ++t) {
    int k0 = wave * 128 + t * 32;
    v16bf ap  = load_a_f32v(S, SLD, k0, lane);       // P -> bf16 fragment
    v16bf bv0 = load_b_t(Vt, VLD, k0, 0, lane);      // B(k,n) = Vt[n][k]
    v16bf bv1 = load_b_t(Vt, VLD, k0, 16, lane);
    o0 = wmma_bf16(ap, bv0, o0);
    o1 = wmma_bf16(ap, bv1, o1);
  }
#pragma unroll
  for (int i = 0; i < 8; ++i) {
    opart[wave * 512 + (rbase + i) * 32 + col]      = o0[i];
    opart[wave * 512 + (rbase + i) * 32 + 16 + col] = o1[i];
  }
  __syncthreads();

  // reduce 8 wave partials, normalize, scatter head block into [N, C]
  for (int idx = tid; idx < 128; idx += 256) {
    int r = idx >> 3, d = (idx & 7) * 4;
    v4f s;
#pragma unroll
    for (int q = 0; q < 4; ++q) s[q] = 0.f;
#pragma unroll
    for (int w = 0; w < 8; ++w) {
      v4f p = *(const v4f*)(opart + w * 512 + r * 32 + d);
#pragma unroll
      for (int q = 0; q < 4; ++q) s[q] += p[q];
    }
    float inv = 1.f / rowsum[r];
#pragma unroll
    for (int q = 0; q < 4; ++q) s[q] *= inv;
    *(v4f*)(attn_o + (rowbase + qt * 16 + r) * Cc + h * 32 + d) = s;
  }
}

// ---------------- Kernel 4: root-linear + out-proj + residual combine ------------
// pre = h_conv + x@W_root + attn_o@Wo + bo + 2*x   (== h1 + h2)
__global__ __launch_bounds__(256) void k_combine(
    const float* __restrict__ x, const float* __restrict__ attn_o,
    const float* __restrict__ h_conv, const __bf16* __restrict__ WbT_root,
    const __bf16* __restrict__ WbT_o, const float* __restrict__ bo,
    float* __restrict__ out, __bf16* __restrict__ pre_bf) {
  __shared__ __bf16 Xs[16 * 136];
  __shared__ __bf16 Os[16 * 136];
  const int tid = threadIdx.x;
  const int m0 = blockIdx.x * 16;
  for (int idx = tid; idx < 16 * 16; idx += 256) {
    int j = idx >> 4, k = (idx & 15) * 8;
    const float* sx = x + (size_t)(m0 + j) * Cc + k;
    const float* so = attn_o + (size_t)(m0 + j) * Cc + k;
    *(v8bf*)(Xs + j * 136 + k) = pack_bf8(*(const v4f*)sx, *(const v4f*)(sx + 4));
    *(v8bf*)(Os + j * 136 + k) = pack_bf8(*(const v4f*)so, *(const v4f*)(so + 4));
  }
  __syncthreads();
  const int wave = tid >> 5, lane = tid & 31;
  const int n0 = wave * 16, col = lane & 15, rbase = (lane >> 4) * 8;
  v8f c;
  float bias = bo[n0 + col];
#pragma unroll
  for (int i = 0; i < 8; ++i) c[i] = bias;
#pragma unroll
  for (int kk = 0; kk < 4; ++kk) {
    v16bf a1 = load_a_bf16v(Xs, 136, kk * 32, lane);
    v16bf b1 = load_b_t(WbT_root, 128, kk * 32, n0, lane);
    c = wmma_bf16(a1, b1, c);
    v16bf a2 = load_a_bf16v(Os, 136, kk * 32, lane);
    v16bf b2 = load_b_t(WbT_o, 128, kk * 32, n0, lane);
    c = wmma_bf16(a2, b2, c);
  }
#pragma unroll
  for (int i = 0; i < 8; ++i) {
    size_t off = (size_t)(m0 + rbase + i) * Cc + n0 + col;
    float v = c[i] + h_conv[off] + 2.f * x[off];
    out[off] = v;
    pre_bf[off] = (__bf16)v;
  }
}

// ---------------- Kernel 5: fused residual MLP  C -> 2C -> C ---------------------
__global__ __launch_bounds__(256) void k_mlp(
    const __bf16* __restrict__ pre_bf, const float* __restrict__ b1,
    const float* __restrict__ b2, const __bf16* __restrict__ WbT1,
    const __bf16* __restrict__ WbT2, float* __restrict__ out) {
  __shared__ __bf16 Ps[16 * 128];
  __shared__ __bf16 Hs[16 * 264];
  const int tid = threadIdx.x;
  const int m0 = blockIdx.x * 16;
  for (int idx = tid; idx < 16 * 16; idx += 256) {
    int j = idx >> 4, k = (idx & 15) * 8;
    *(v8bf*)(Ps + j * 128 + k) =
        *(const v8bf*)(pre_bf + (size_t)(m0 + j) * Cc + k);
  }
  __syncthreads();
  const int wave = tid >> 5, lane = tid & 31;
  const int col = lane & 15, rbase = (lane >> 4) * 8;
  // stage 1: hidden = relu(pre @ W1 + b1), 16x256, two 16-col tiles per wave
#pragma unroll
  for (int t = 0; t < 2; ++t) {
    int n0 = wave * 32 + t * 16;
    v8f c;
    float bias = b1[n0 + col];
#pragma unroll
    for (int i = 0; i < 8; ++i) c[i] = bias;
#pragma unroll
    for (int kk = 0; kk < 4; ++kk) {
      v16bf a = load_a_bf16v(Ps, 128, kk * 32, lane);
      v16bf b = load_b_t(WbT1, 128, kk * 32, n0, lane);
      c = wmma_bf16(a, b, c);
    }
#pragma unroll
    for (int i = 0; i < 8; ++i)
      Hs[(rbase + i) * 264 + n0 + col] = (__bf16)fmaxf(c[i], 0.f);
  }
  __syncthreads();
  // stage 2: out += hidden @ W2 + b2
  {
    int n0 = wave * 16;
    v8f c;
    float bias = b2[n0 + col];
#pragma unroll
    for (int i = 0; i < 8; ++i) c[i] = bias;
#pragma unroll
    for (int kk = 0; kk < 8; ++kk) {
      v16bf a = load_a_bf16v(Hs, 264, kk * 32, lane);
      v16bf b = load_b_t(WbT2, 256, kk * 32, n0, lane);
      c = wmma_bf16(a, b, c);
    }
#pragma unroll
    for (int i = 0; i < 8; ++i) {
      size_t off = (size_t)(m0 + rbase + i) * Cc + n0 + col;
      out[off] += c[i];
    }
  }
}

// ---------------- workspace layout (weights stored K-major / transposed) ---------
constexpr size_t OFF_WEDGE = 0;                                   // [128][384] bf16
constexpr size_t OFF_WROOT = OFF_WEDGE + (size_t)384 * 128 * 2;   // [128][128]
constexpr size_t OFF_WQKV  = OFF_WROOT + (size_t)128 * 128 * 2;   // [384][128]
constexpr size_t OFF_WO    = OFF_WQKV + (size_t)128 * 384 * 2;    // [128][128]
constexpr size_t OFF_W1    = OFF_WO + (size_t)128 * 128 * 2;      // [256][128]
constexpr size_t OFF_W2    = OFF_W1 + (size_t)128 * 256 * 2;      // [128][256]
constexpr size_t OFF_HCONV = 393216;                              // N*C f32
constexpr size_t OFF_QKV   = OFF_HCONV + (size_t)Nn * Cc * 4;     // N*384 bf16
constexpr size_t OFF_AO    = OFF_QKV + (size_t)Nn * 384 * 2;      // N*C f32
constexpr size_t OFF_PRE   = OFF_AO + (size_t)Nn * Cc * 4;        // N*C bf16

extern "C" void kernel_launch(void* const* d_in, const int* in_sizes, int n_in,
                              void* d_out, int out_size, void* d_ws, size_t ws_size,
                              hipStream_t stream) {
  (void)in_sizes; (void)n_in; (void)out_size; (void)ws_size;
  const float* x      = (const float*)d_in[0];
  const int*   eidx   = (const int*)d_in[1];
  const float* eattr  = (const float*)d_in[2];
  /* d_in[3] = batch (unused: equal-size graphs) */
  const float* W_edge = (const float*)d_in[4];
  const float* b_edge = (const float*)d_in[5];
  const float* W_root = (const float*)d_in[6];
  const float* Wqkv   = (const float*)d_in[7];
  const float* bqkv   = (const float*)d_in[8];
  const float* Wo     = (const float*)d_in[9];
  const float* bo     = (const float*)d_in[10];
  const float* W1     = (const float*)d_in[11];
  const float* b1     = (const float*)d_in[12];
  const float* W2     = (const float*)d_in[13];
  const float* b2     = (const float*)d_in[14];

  char* ws = (char*)d_ws;
  __bf16* WbT_edge = (__bf16*)(ws + OFF_WEDGE);
  __bf16* WbT_root = (__bf16*)(ws + OFF_WROOT);
  __bf16* WbT_qkv  = (__bf16*)(ws + OFF_WQKV);
  __bf16* WbT_o    = (__bf16*)(ws + OFF_WO);
  __bf16* WbT_1    = (__bf16*)(ws + OFF_W1);
  __bf16* WbT_2    = (__bf16*)(ws + OFF_W2);
  float*  h_conv   = (float*)(ws + OFF_HCONV);
  __bf16* qkv      = (__bf16*)(ws + OFF_QKV);
  float*  attn_o   = (float*)(ws + OFF_AO);
  __bf16* pre_bf   = (__bf16*)(ws + OFF_PRE);

  float* out   = (float*)d_out;
  float* e_new = out + (size_t)Nn * Cc;   // tuple output: (out, e_new) flat-concat

  auto cvtT = [&](const float* s, __bf16* d, int K, int Ncol) {
    int n = K * Ncol;
    k_cvt_t<<<(n + 255) / 256, 256, 0, stream>>>(s, d, K, Ncol);
  };
  cvtT(W_edge, WbT_edge, 384, 128);
  cvtT(W_root, WbT_root, 128, 128);
  cvtT(Wqkv,   WbT_qkv,  128, 384);
  cvtT(Wo,     WbT_o,    128, 128);
  cvtT(W1,     WbT_1,    128, 256);
  cvtT(W2,     WbT_2,    256, 128);

  hipMemsetAsync(h_conv, 0, (size_t)Nn * Cc * sizeof(float), stream);

  k_edge_mlp<<<Ee / 16, 256, 0, stream>>>(x, eidx, eattr, b_edge, WbT_edge,
                                          e_new, h_conv);
  k_qkv<<<dim3(Nn / 16, 3), 256, 0, stream>>>(x, bqkv, WbT_qkv, qkv);

  hipFuncSetAttribute((const void*)k_attn,
                      hipFuncAttributeMaxDynamicSharedMemorySize, (int)A_TOT);
  k_attn<<<32 * 4 * (Ll / 16), 256, A_TOT, stream>>>(qkv, attn_o);

  k_combine<<<Nn / 16, 256, 0, stream>>>(x, attn_o, h_conv, WbT_root, WbT_o, bo,
                                         out, pre_bf);
  k_mlp<<<Nn / 16, 256, 0, stream>>>(pre_bf, b1, b2, WbT_1, WbT_2, out);
}
